// SpanAttention_5995774345596
// MI455X (gfx1250) — compile-verified, matrix operand order checked by reading, not run on hardware
//
#include <hip/hip_runtime.h>
#include <hip/hip_bf16.h>

typedef __bf16 bf16;
typedef __attribute__((ext_vector_type(8)))  bf16  v8bf;
typedef __attribute__((ext_vector_type(16))) bf16  v16bf;
typedef __attribute__((ext_vector_type(8)))  float v8f;

#define B_   8
#define L_   512
#define D_   1024
#define S_   6144
#define M_   (B_ * S_)   /* 49152 spans total */
#define KPAD 40          /* LDS row stride in bf16 (32 + 8 pad), 80 bytes */

// ---------------------------------------------------------------------------
// Kernel 1: exclusive-prefix-sum of h along L into cs [B, L+1, D].
// One thread per (b, d) column; loads/stores coalesced across d.
// ---------------------------------------------------------------------------
__global__ __launch_bounds__(256) void span_cumsum_kernel(
    const float* __restrict__ h, float* __restrict__ cs) {
  int idx = blockIdx.x * blockDim.x + threadIdx.x;   // 0 .. B*D-1
  int b = idx >> 10;          // / D_
  int d = idx & (D_ - 1);
  const float* hp = h  + (size_t)b * L_ * D_ + d;
  float*       cp = cs + (size_t)b * (L_ + 1) * D_ + d;
  float acc = 0.0f;
  cp[0] = 0.0f;
  for (int l = 0; l < L_; ++l) {
    acc += hp[(size_t)l * D_];
    cp[(size_t)(l + 1) * D_] = acc;
  }
}

// ---------------------------------------------------------------------------
// Kernel 2: fused span-mean + GEMM + bias + ReLU, double-buffered LDS.
// Block = 256 threads = 8 waves (wave32). Output tile 128(M) x 128(N).
// Wave grid 4x2: each wave owns 32(M) x 64(N) => 2x4 wmma accumulators.
// Pipeline per 32-wide K step:
//   issue global loads for step+1 -> frag loads + 8 WMMAs from LDS[buf]
//   -> convert/store step+1 into LDS[buf^1] -> single barrier.
// ---------------------------------------------------------------------------
__global__ __launch_bounds__(256) void span_gemm_kernel(
    const float* __restrict__ cs,
    const long long* __restrict__ span_idx,
    const float* __restrict__ W,
    const float* __restrict__ bias,
    float* __restrict__ out) {

  __shared__ __align__(16) bf16 sA[2][128 * KPAD];
  __shared__ __align__(16) bf16 sB[2][128 * KPAD];
  __shared__ int   sOffE[128];
  __shared__ int   sOffS[128];
  __shared__ float sInv[128];

  const int tid   = threadIdx.x;
  const int mbase = blockIdx.y * 128;   // global span-row base
  const int nbase = blockIdx.x * 128;   // output-feature base

  // Per-span metadata (start/end row offsets into cs, 1/count or 0)
  if (tid < 128) {
    int g = mbase + tid;
    int b = g / S_;
    int s = g - b * S_;
    long long s0 = span_idx[((size_t)(b * S_ + s)) * 2 + 0];
    long long s1 = span_idx[((size_t)(b * S_ + s)) * 2 + 1];
    int st = (int)(s0 < 0 ? 0 : (s0 > (L_ - 1) ? (L_ - 1) : s0));
    int en = (int)(s1 < 0 ? 0 : (s1 > (L_ - 1) ? (L_ - 1) : s1));
    bool valid = (st <= en);
    sOffE[tid] = (b * (L_ + 1) + en + 1) * D_;
    sOffS[tid] = (b * (L_ + 1) + st) * D_;
    sInv[tid]  = valid ? (1.0f / (float)(en - st + 1)) : 0.0f;
  }
  __syncthreads();

  const int lane  = tid & 31;
  const int wave  = tid >> 5;      // 0..7
  const int wm    = wave >> 1;     // 0..3 : 32-row M sub-block
  const int wn    = wave & 1;      // 0..1 : 64-col N sub-block
  const int lrow  = lane & 15;
  const int lhalf = lane >> 4;     // 0 or 1

  v8f acc[2][4];
#pragma unroll
  for (int a = 0; a < 2; ++a)
#pragma unroll
    for (int n = 0; n < 4; ++n) {
      v8f z = {};
      acc[a][n] = z;
    }

  // Staging role: each thread owns 16 bf16 of one row per tile per K step.
  const int r    = tid >> 1;        // row 0..127
  const int doff = (tid & 1) << 4;  // 16-wide half of the 32-wide K chunk

  const int   offE = sOffE[r];
  const int   offS = sOffS[r];
  const float inv  = sInv[r];
  const float* wRow = W + (size_t)(nbase + r) * D_ + doff;

  float4 ae[4], as4[4], wv[4];   // prefetch registers (12 x float4)

  auto loadChunk = [&](int kk) {
    const float4* pe = (const float4*)(cs + offE + kk + doff);
    const float4* ps = (const float4*)(cs + offS + kk + doff);
    const float4* pw = (const float4*)(wRow + kk);
#pragma unroll
    for (int q = 0; q < 4; ++q) {
      ae[q]  = pe[q];
      as4[q] = ps[q];
      wv[q]  = pw[q];
    }
  };

  auto storeChunk = [&](int buf) {
    v8bf alo, ahi, wlo, whi;
#pragma unroll
    for (int q = 0; q < 2; ++q) {
      float4 e = ae[q], sv = as4[q], w = wv[q];
      alo[4 * q + 0] = (bf16)((e.x - sv.x) * inv);
      alo[4 * q + 1] = (bf16)((e.y - sv.y) * inv);
      alo[4 * q + 2] = (bf16)((e.z - sv.z) * inv);
      alo[4 * q + 3] = (bf16)((e.w - sv.w) * inv);
      wlo[4 * q + 0] = (bf16)w.x;
      wlo[4 * q + 1] = (bf16)w.y;
      wlo[4 * q + 2] = (bf16)w.z;
      wlo[4 * q + 3] = (bf16)w.w;
    }
#pragma unroll
    for (int q = 2; q < 4; ++q) {
      float4 e = ae[q], sv = as4[q], w = wv[q];
      ahi[4 * (q - 2) + 0] = (bf16)((e.x - sv.x) * inv);
      ahi[4 * (q - 2) + 1] = (bf16)((e.y - sv.y) * inv);
      ahi[4 * (q - 2) + 2] = (bf16)((e.z - sv.z) * inv);
      ahi[4 * (q - 2) + 3] = (bf16)((e.w - sv.w) * inv);
      whi[4 * (q - 2) + 0] = (bf16)w.x;
      whi[4 * (q - 2) + 1] = (bf16)w.y;
      whi[4 * (q - 2) + 2] = (bf16)w.z;
      whi[4 * (q - 2) + 3] = (bf16)w.w;
    }
    bf16* da = &sA[buf][r * KPAD + doff];
    *(v8bf*)(da)     = alo;
    *(v8bf*)(da + 8) = ahi;
    bf16* db = &sB[buf][r * KPAD + doff];
    *(v8bf*)(db)     = wlo;
    *(v8bf*)(db + 8) = whi;
  };

  // Prologue: stage chunk 0 into buffer 0.
  loadChunk(0);
  storeChunk(0);
  __syncthreads();

  for (int kk = 0; kk < D_; kk += 32) {
    const int  buf      = (kk >> 5) & 1;
    const bool has_next = (kk + 32) < D_;

    // Issue next chunk's global loads early (overlap with WMMAs below).
    if (has_next) loadChunk(kk + 32);

    // Fragments: two contiguous 16B LDS chunks per lane.
    // 16-bit A layout: lane<16 holds K {0..7,16..23}; lane>=16 holds K {8..15,24..31}
    v16bf afrag[2], bfrag[4];
#pragma unroll
    for (int a = 0; a < 2; ++a) {
      const bf16* ap = &sA[buf][(wm * 32 + a * 16 + lrow) * KPAD + 8 * lhalf];
      v8bf c0 = *(const v8bf*)(ap);
      v8bf c1 = *(const v8bf*)(ap + 16);
      afrag[a] = __builtin_shufflevector(c0, c1, 0, 1, 2, 3, 4, 5, 6, 7,
                                         8, 9, 10, 11, 12, 13, 14, 15);
    }
#pragma unroll
    for (int n = 0; n < 4; ++n) {
      const bf16* bp = &sB[buf][(wn * 64 + n * 16 + lrow) * KPAD + 8 * lhalf];
      v8bf c0 = *(const v8bf*)(bp);
      v8bf c1 = *(const v8bf*)(bp + 16);
      bfrag[n] = __builtin_shufflevector(c0, c1, 0, 1, 2, 3, 4, 5, 6, 7,
                                         8, 9, 10, 11, 12, 13, 14, 15);
    }

#pragma unroll
    for (int a = 0; a < 2; ++a)
#pragma unroll
      for (int n = 0; n < 4; ++n)
        acc[a][n] = __builtin_amdgcn_wmma_f32_16x16x32_bf16(
            false, afrag[a], false, bfrag[n], (short)0, acc[a][n],
            false, false);

    // Stage next chunk into the other buffer; single barrier per step.
    if (has_next) storeChunk(buf ^ 1);
    __syncthreads();
  }

  // ---- epilogue: bias + ReLU, non-temporal fp32 stores ----
  // C/D layout: vgpr v, lane -> M = v + 8*(lane>=16), N = lane&15
#pragma unroll
  for (int n = 0; n < 4; ++n) {
    int col = nbase + wn * 64 + n * 16 + lrow;
    float bv = bias[col];
#pragma unroll
    for (int a = 0; a < 2; ++a) {
      int mrow0 = mbase + wm * 32 + a * 16 + 8 * lhalf;
#pragma unroll
      for (int v = 0; v < 8; ++v) {
        float x = acc[a][n][v] + bv;
        x = x > 0.0f ? x : 0.0f;
        __builtin_nontemporal_store(x, &out[(size_t)(mrow0 + v) * D_ + col]);
      }
    }
  }
}

// ---------------------------------------------------------------------------
extern "C" void kernel_launch(void* const* d_in, const int* in_sizes, int n_in,
                              void* d_out, int out_size, void* d_ws, size_t ws_size,
                              hipStream_t stream) {
  (void)in_sizes; (void)n_in; (void)out_size; (void)ws_size;
  const float*     h        = (const float*)d_in[0];
  const long long* span_idx = (const long long*)d_in[1];
  const float*     W        = (const float*)d_in[2];
  const float*     bias     = (const float*)d_in[3];
  float*           out      = (float*)d_out;
  float*           cs       = (float*)d_ws;   // B*(L+1)*D floats = 16.8 MB

  span_cumsum_kernel<<<dim3((B_ * D_) / 256), dim3(256), 0, stream>>>(h, cs);
  span_gemm_kernel<<<dim3(D_ / 128, M_ / 128), dim3(256), 0, stream>>>(
      cs, span_idx, W, bias, out);
}
